// Model1_52656299049486
// MI455X (gfx1250) — matrix operand order repaired and checked by reading, not compiled, not run on hardware
//
#include <hip/hip_runtime.h>
#include <hip/hip_bf16.h>

// ---------------------------------------------------------------------------
// GNN message-passing pipeline for MI455X (gfx1250, wave32, WMMA).
//
// Stage 1 (edge): per 16-edge tile, A[16x48->16x64 f16] x B[64x16 f16(W_mpl
//   padded)] via two v_wmma_f32_16x16x32_f16, relu+bias, f32 atomic scatter
//   to x_nodes[N,10].  Memory-bound (~230MB HBM stream, node table L2-hot);
//   WMMA makes the MLP free.
// Stage 2 (node): x_nodes @ W1 + b1, relu, atomic scatter by batch -> g_sums.
// Stage 3 (graph): relu(g @ W2 + b2) @ W3 + b3 -> out[4096].
// ---------------------------------------------------------------------------

typedef __attribute__((ext_vector_type(16))) _Float16 v16h;
typedef __attribute__((ext_vector_type(8)))  float    v8f;

#define GNN_N_NODES  100000
#define GNN_N_EDGES  3200000
#define GNN_D_FEAT   16
#define GNN_D_IN     48          // 3 * D_FEAT
#define GNN_D_MSG    10
#define GNN_N_GRAPHS 4096

__global__ void gnn_zero_kernel(float* __restrict__ p, int n) {
    int i = blockIdx.x * blockDim.x + threadIdx.x;
    if (i < n) p[i] = 0.0f;
}

__global__ __launch_bounds__(256)
void gnn_edge_wmma_kernel(const int*   __restrict__ ei,        // [2, E]
                          const float* __restrict__ node_attr, // [N, 16]
                          const float* __restrict__ edge_attr, // [E, 16]
                          const float* __restrict__ W_mpl,     // [48, 10]
                          const float* __restrict__ b_mpl,     // [10]
                          float*       __restrict__ x_nodes)   // [N, 10] accum
{
    const int lane  = threadIdx.x & 31;
    const int m     = lane & 15;   // A row within tile == C column index n
    const int kh    = lane >> 4;   // K-half selector (A) / K-group (B)
    const int waveI = threadIdx.x >> 5;
    const int wavesPerBlock = blockDim.x >> 5;
    const int gwave  = blockIdx.x * wavesPerBlock + waveI;
    const int nwaves = gridDim.x * wavesPerBlock;

    // ---- Build B (weights, K x N = 64 x 16, zero-padded) once per wave ----
    // B layout (16-bit, wave32): lane n = l&15 holds column n; element t holds
    // K = 8*(l>>4) + t (t<8), 16 + 8*(l>>4) + t-8 (t>=8).  WMMA0: K=feature 0..31, WMMA1: 32..63.
    const int n = m; // this lane's B / C column
    v16h Bw0, Bw1;
    #pragma unroll
    for (int t = 0; t < 16; ++t) {
        int k0 = (t < 8) ? (8 * kh + t) : (16 + 8 * kh + (t - 8)); // feature index for WMMA0 (<32, valid)
        int k1 = 32 + k0;                    // feature index for WMMA1
        float w0 = (n < GNN_D_MSG) ? W_mpl[k0 * GNN_D_MSG + n] : 0.0f;
        float w1 = (n < GNN_D_MSG && k1 < GNN_D_IN) ? W_mpl[k1 * GNN_D_MSG + n] : 0.0f;
        Bw0[t] = (_Float16)w0;
        Bw1[t] = (_Float16)w1;
    }
    const float bias = (n < GNN_D_MSG) ? b_mpl[n] : 0.0f;

    const int numTiles = (GNN_N_EDGES + 15) / 16;
    for (int tile = gwave; tile < numTiles; tile += nwaves) {
        const int e     = tile * 16 + m;
        const bool valid = (e < GNN_N_EDGES);
        const int ec    = valid ? e : (GNN_N_EDGES - 1);
        const int src   = ei[ec];
        const int dst   = ei[GNN_N_EDGES + ec];

        // ---- Build A (16 edges x 64 f16).  Lane (m, kh) holds row m:
        //   WMMA0: v[0..7]=x_src[8kh..8kh+7], v[8..15]=x_dst[8kh..8kh+7]
        //   WMMA1: v[0..7]=edge_attr[8kh..8kh+7], v[8..15]=0 (pad K 48..63)
        const float4* ps = (const float4*)(node_attr + (size_t)src * GNN_D_FEAT + 8 * kh);
        const float4* pd = (const float4*)(node_attr + (size_t)dst * GNN_D_FEAT + 8 * kh);
        const float4* pe = (const float4*)(edge_attr + (size_t)ec  * GNN_D_FEAT + 8 * kh);
        float4 s0 = ps[0], s1 = ps[1];
        float4 d0 = pd[0], d1 = pd[1];
        float4 e0 = pe[0], e1 = pe[1];
        const float msk = valid ? 1.0f : 0.0f;

        v16h A0, A1;
        A0[0]=(_Float16)(s0.x*msk); A0[1]=(_Float16)(s0.y*msk);
        A0[2]=(_Float16)(s0.z*msk); A0[3]=(_Float16)(s0.w*msk);
        A0[4]=(_Float16)(s1.x*msk); A0[5]=(_Float16)(s1.y*msk);
        A0[6]=(_Float16)(s1.z*msk); A0[7]=(_Float16)(s1.w*msk);
        A0[8]=(_Float16)(d0.x*msk); A0[9]=(_Float16)(d0.y*msk);
        A0[10]=(_Float16)(d0.z*msk); A0[11]=(_Float16)(d0.w*msk);
        A0[12]=(_Float16)(d1.x*msk); A0[13]=(_Float16)(d1.y*msk);
        A0[14]=(_Float16)(d1.z*msk); A0[15]=(_Float16)(d1.w*msk);

        A1[0]=(_Float16)(e0.x*msk); A1[1]=(_Float16)(e0.y*msk);
        A1[2]=(_Float16)(e0.z*msk); A1[3]=(_Float16)(e0.w*msk);
        A1[4]=(_Float16)(e1.x*msk); A1[5]=(_Float16)(e1.y*msk);
        A1[6]=(_Float16)(e1.z*msk); A1[7]=(_Float16)(e1.w*msk);
        #pragma unroll
        for (int t = 8; t < 16; ++t) A1[t] = (_Float16)0.0f;

        // ---- D = A0*B0 + A1*B1  (f32 accumulate) ----
        v8f c = {};
        c = __builtin_amdgcn_wmma_f32_16x16x32_f16(false, A0, false, Bw0,
                                                   (short)0, c, false, false);
        c = __builtin_amdgcn_wmma_f32_16x16x32_f16(false, A1, false, Bw1,
                                                   (short)0, c, false, false);

        // dst index of every row this lane covers (rows 8*kh + r); hoist the
        // shuffles before divergent code: source lane `row` (<16) holds edge
        // tile*16+row's dst.
        int drs[8];
        #pragma unroll
        for (int r = 0; r < 8; ++r) drs[r] = __shfl(dst, 8 * kh + r, 32);

        // ---- C layout: lane l, vgpr r -> row 8*(l>>4)+r, col l&15 ----
        if (n < GNN_D_MSG) {
            #pragma unroll
            for (int r = 0; r < 8; ++r) {
                int row = 8 * kh + r;
                int er  = tile * 16 + row;
                if (er < GNN_N_EDGES) {
                    float v = c[r] + bias;
                    v = v > 0.0f ? v : 0.0f;
                    atomicAdd(&x_nodes[(size_t)drs[r] * GNN_D_MSG + n], v);
                }
            }
        }
    }
}

__global__ void gnn_node_kernel(const float* __restrict__ x_nodes, // [N,10]
                                const int*   __restrict__ batch,   // [N]
                                const float* __restrict__ W1,      // [10,5]
                                const float* __restrict__ b1,      // [5]
                                float*       __restrict__ g_sums)  // [G,5] accum
{
    int i = blockIdx.x * blockDim.x + threadIdx.x;
    if (i >= GNN_N_NODES) return;
    float xin[GNN_D_MSG];
    #pragma unroll
    for (int j = 0; j < GNN_D_MSG; ++j) xin[j] = x_nodes[(size_t)i * GNN_D_MSG + j];
    int g = batch[i];
    #pragma unroll
    for (int o = 0; o < 5; ++o) {
        float acc = b1[o];
        #pragma unroll
        for (int j = 0; j < GNN_D_MSG; ++j) acc += xin[j] * W1[j * 5 + o];
        acc = acc > 0.0f ? acc : 0.0f;
        atomicAdd(&g_sums[g * 5 + o], acc);
    }
}

__global__ void gnn_graph_kernel(const float* __restrict__ g_sums, // [G,5]
                                 const float* __restrict__ W2,     // [5,5]
                                 const float* __restrict__ b2,     // [5]
                                 const float* __restrict__ W3,     // [5,1]
                                 const float* __restrict__ b3,     // [1]
                                 float*       __restrict__ out)    // [G]
{
    int g = blockIdx.x * blockDim.x + threadIdx.x;
    if (g >= GNN_N_GRAPHS) return;
    float xin[5], h[5];
    #pragma unroll
    for (int j = 0; j < 5; ++j) xin[j] = g_sums[g * 5 + j];
    #pragma unroll
    for (int o = 0; o < 5; ++o) {
        float acc = b2[o];
        #pragma unroll
        for (int j = 0; j < 5; ++j) acc += xin[j] * W2[j * 5 + o];
        h[o] = acc > 0.0f ? acc : 0.0f;
    }
    float y = b3[0];
    #pragma unroll
    for (int j = 0; j < 5; ++j) y += h[j] * W3[j];
    out[g] = y;
}

extern "C" void kernel_launch(void* const* d_in, const int* in_sizes, int n_in,
                              void* d_out, int out_size, void* d_ws, size_t ws_size,
                              hipStream_t stream) {
    const int*   ei        = (const int*)  d_in[0];   // edge_index [2,E]
    const float* node_attr = (const float*)d_in[1];   // [N,16]
    const float* edge_attr = (const float*)d_in[2];   // [E,16]
    const int*   batch     = (const int*)  d_in[3];   // [N]
    const float* W_mpl     = (const float*)d_in[4];   // [48,10]
    const float* b_mpl     = (const float*)d_in[5];   // [10]
    const float* W1        = (const float*)d_in[6];   // [10,5]
    const float* b1        = (const float*)d_in[7];   // [5]
    const float* W2        = (const float*)d_in[8];   // [5,5]
    const float* b2        = (const float*)d_in[9];   // [5]
    const float* W3        = (const float*)d_in[10];  // [5,1]
    const float* b3        = (const float*)d_in[11];  // [1]
    float* out = (float*)d_out;

    // Workspace: x_nodes [N,10] then g_sums [G,5] (contiguous, zeroed per call)
    float* x_nodes = (float*)d_ws;
    float* g_sums  = x_nodes + (size_t)GNN_N_NODES * GNN_D_MSG;
    const int nZero = GNN_N_NODES * GNN_D_MSG + GNN_N_GRAPHS * 5;

    gnn_zero_kernel<<<(nZero + 255) / 256, 256, 0, stream>>>(x_nodes, nZero);

    // 200K 16-edge tiles over 2048 blocks x 8 waves = 16384 waves (~12 tiles/wave)
    gnn_edge_wmma_kernel<<<2048, 256, 0, stream>>>(ei, node_attr, edge_attr,
                                                   W_mpl, b_mpl, x_nodes);

    gnn_node_kernel<<<(GNN_N_NODES + 255) / 256, 256, 0, stream>>>(
        x_nodes, batch, W1, b1, g_sums);

    gnn_graph_kernel<<<(GNN_N_GRAPHS + 255) / 256, 256, 0, stream>>>(
        g_sums, W2, b2, W3, b3, out);
}
